// VectorQuantization_16432544874769
// MI455X (gfx1250) — matrix operand order repaired and checked by reading, not compiled, not run on hardware
//
#include <hip/hip_runtime.h>
#include <math.h>

// ---------------------------------------------------------------------------
// VectorQuantization for MI455X (gfx1250, wave32, WMMA + async-to-LDS DMA,
// double-buffered B tiles)
// x: [8192, 512] f32, codebook: [16384, 512] f32
// out: codebook[argmax_k (normalize(x) @ codebook.T)]  -> [8192, 512] f32
// ---------------------------------------------------------------------------

constexpr int N_ROWS  = 8192;
constexpr int DIM     = 512;
constexpr int K_CODES = 16384;

constexpr int ROWS_PER_BLOCK  = 256;     // 8 waves x 32-row (2-tile) slices
constexpr int LDS_STRIDE      = 520;     // A: 512 + 8 pad (bank spread, 16B align)
constexpr int B_STRIDE_Q      = 136;     // B: 128 + 8 pad per D-quarter
constexpr int CODE_CHUNKS     = 16;      // grid.y split over codes
constexpr int CODES_PER_CHUNK = K_CODES / CODE_CHUNKS;        // 1024
constexpr int CODES_PER_ITER  = 64;      // 4 code-tiles per wave iteration
constexpr int ITERS_PER_CHUNK = CODES_PER_CHUNK / CODES_PER_ITER;  // 16

typedef __attribute__((ext_vector_type(16))) __bf16       v16bf;
typedef __attribute__((ext_vector_type(8)))  float        v8f;
typedef __attribute__((ext_vector_type(4)))  unsigned int v4u;
typedef __attribute__((ext_vector_type(2)))  unsigned int v2u;
typedef __attribute__((ext_vector_type(4)))  float        v4f;
typedef __attribute__((ext_vector_type(2)))  float        v2f;

union FragBF { v16bf bf; v4u u[2]; };

__device__ __forceinline__ unsigned int f2bf(float f) {
  // round-to-nearest-even fp32 -> bf16
  unsigned int u = __float_as_uint(f);
  return (u + 0x7FFFu + ((u >> 16) & 1u)) >> 16;
}

// LDS byte offset of a generic pointer derived from shared memory
// (ptrtoint on an addrspace(3) pointer yields the raw LDS offset).
__device__ __forceinline__ unsigned lds_off_of(const void* p) {
  return (unsigned)(unsigned long long)(const __attribute__((address_space(3))) char*)p;
}

// CDNA5 async DMA: 16B global -> LDS, tracked with ASYNCcnt (no VGPR data).
__device__ __forceinline__ void async_copy16(unsigned lds_off, const void* gaddr) {
  asm volatile("global_load_async_to_lds_b128 %0, %1, off"
               :
               : "v"(lds_off), "v"(gaddr)
               : "memory");
}

__device__ __forceinline__ void wait_async0() {
  asm volatile("s_wait_asynccnt 0x0" ::: "memory");
}

// Stage one D-quarter of a 64-code B tile (16 KB) via async DMA.
__device__ __forceinline__ void stage_b_phase(
    const unsigned short* __restrict__ cbq, unsigned short* Bbuf,
    int cBase, int q, int tr, int tcq) {
#pragma unroll
  for (int rr = 0; rr < 4; ++rr) {
    const int r = tr + rr * 16;
    async_copy16(lds_off_of(Bbuf + r * B_STRIDE_Q + tcq),
                 cbq + (size_t)(cBase + r) * DIM + q * 128 + tcq);
  }
}

// --------------------------- kernel 1: normalize ---------------------------
__global__ __launch_bounds__(256) void vq_normalize_bf16(
    const float* __restrict__ x, unsigned short* __restrict__ xq) {
  __shared__ float red[8];
  const int row = blockIdx.x;
  const int t   = threadIdx.x;
  v2f v = ((const v2f*)(x + (size_t)row * DIM))[t];
  float s = v.x * v.x + v.y * v.y;
#pragma unroll
  for (int off = 16; off >= 1; off >>= 1) s += __shfl_xor(s, off, 32);
  if ((t & 31) == 0) red[t >> 5] = s;
  __syncthreads();
  float tot = 0.0f;
#pragma unroll
  for (int i = 0; i < 8; ++i) tot += red[i];
  const float inv = 1.0f / fmaxf(sqrtf(tot), 1e-12f);
  const unsigned int packed = f2bf(v.x * inv) | (f2bf(v.y * inv) << 16);
  ((unsigned int*)xq)[(size_t)row * (DIM / 2) + t] = packed;
}

// ----------------------- kernel 2: codebook -> bf16 ------------------------
__global__ __launch_bounds__(256) void vq_cvt_bf16(
    const float* __restrict__ cb, unsigned short* __restrict__ cbq) {
  const size_t i = ((size_t)blockIdx.x * 256 + threadIdx.x) * 4;
  v4f v = *(const v4f*)(cb + i);
  v2u o;
  o.x = f2bf(v.x) | (f2bf(v.y) << 16);
  o.y = f2bf(v.z) | (f2bf(v.w) << 16);
  *(v2u*)(cbq + i) = o;
}

// ------------------- kernel 3: WMMA similarities + argmax ------------------
// grid = (N_ROWS/256, CODE_CHUNKS), block = 256 (8 waves).
// A tile (256x512 bf16) DMA'd into LDS once; 64-code B tiles stream through
// LDS in 4 D-quarter phases with two buffers: phase q+1's DMA is issued right
// after the barrier and overlaps phase q's compute. Each wave: 2 row-tiles x
// 4 code-tiles -> 8 WMMAs per K-step from 12 ds_load_b128 (1.5 loads/WMMA).
__global__ __launch_bounds__(256) void vq_sim_argmax(
    const unsigned short* __restrict__ xq,
    const unsigned short* __restrict__ cbq,
    v2f* __restrict__ partial) {
  extern __shared__ unsigned char smem[];
  unsigned short* As  = (unsigned short*)smem;                     // 256 x 520
  unsigned short* Bs0 = As + ROWS_PER_BLOCK * LDS_STRIDE;          // 64 x 136
  unsigned short* Bs1 = Bs0 + 64 * B_STRIDE_Q;                     // 64 x 136
  float*          Val = (float*)(Bs1 + 64 * B_STRIDE_Q);           // 256
  int*            Idx = (int*)(Val + ROWS_PER_BLOCK);              // 256

  const int t   = threadIdx.x;
  const int tr  = t >> 4;           // 0..15 staging row
  const int tc  = (t & 15) * 32;    // A staging: 32-element column chunk
  const int tcq = (t & 15) * 8;     // B staging: 8-element column chunk
  const int rowBase = blockIdx.x * ROWS_PER_BLOCK;

  // Stage A tile (256 rows x 512 bf16) via async DMA
  for (int r = tr; r < ROWS_PER_BLOCK; r += 16) {
    const unsigned short* src = xq + (size_t)(rowBase + r) * DIM + tc;
    const unsigned dst = lds_off_of(As + r * LDS_STRIDE + tc);
    async_copy16(dst,      src);
    async_copy16(dst + 16, src + 8);
    async_copy16(dst + 32, src + 16);
    async_copy16(dst + 48, src + 24);
  }

  const int wave = t >> 5;
  const int lane = t & 31;
  const int hl   = lane >> 4;       // half of wave
  const int ln   = lane & 15;
  // A frags: rows wave*32+ln (tile0) and +16 (tile1);
  // K slots {0..7,16..23} (lo half) / {8..15,24..31} (hi half)
  const unsigned short* aBase0 = As + (wave * 32 + ln) * LDS_STRIDE + hl * 8;
  const unsigned short* aBase1 = aBase0 + 16 * LDS_STRIDE;

  float bestv0[8], bestv1[8];
  int   besti0[8], besti1[8];
#pragma unroll
  for (int r = 0; r < 8; ++r) {
    bestv0[r] = -3.402823466e38f; besti0[r] = 0;
    bestv1[r] = -3.402823466e38f; besti1[r] = 0;
  }

  const int chunkBase = blockIdx.y * CODES_PER_CHUNK;

  // Prologue: DMA first B phase (ct=0, q=0) into buffer 0.
  stage_b_phase(cbq, Bs0, chunkBase, 0, tr, tcq);

  for (int ct = 0; ct < ITERS_PER_CHUNK; ++ct) {
    const int codeBase = chunkBase + ct * CODES_PER_ITER;

    v8f acc[2][4];
#pragma unroll
    for (int rt = 0; rt < 2; ++rt)
#pragma unroll
      for (int j = 0; j < 4; ++j) acc[rt][j] = {};

#pragma unroll
    for (int q = 0; q < 4; ++q) {
      wait_async0();     // phase-q DMA (and A tile on the very first pass) done
      __syncthreads();   // all waves' DMA visible; old buffer fully consumed

      // Kick off DMA for the next phase into the other buffer; it overlaps
      // this phase's compute.
      {
        const int nq  = (q + 1) & 3;
        const int nct = (q == 3) ? ct + 1 : ct;
        if (nct < ITERS_PER_CHUNK) {
          unsigned short* nB = (q & 1) ? Bs0 : Bs1;   // buffer (q+1)&1
          stage_b_phase(cbq, nB, chunkBase + nct * CODES_PER_ITER, nq, tr, tcq);
        }
      }

      // Compute 4 K-steps (D elements q*128 .. q*128+127) from buffer q&1.
      const unsigned short* bb =
          ((q & 1) ? Bs1 : Bs0) + ln * B_STRIDE_Q + hl * 8;
#pragma unroll
      for (int ks = 0; ks < 4; ++ks) {
        const int ksg = q * 4 + ks;          // A uses global K-step
        FragBF a[2], b[4];
        a[0].u[0] = *(const v4u*)(aBase0 + ksg * 32);
        a[0].u[1] = *(const v4u*)(aBase0 + ksg * 32 + 16);
        a[1].u[0] = *(const v4u*)(aBase1 + ksg * 32);
        a[1].u[1] = *(const v4u*)(aBase1 + ksg * 32 + 16);
#pragma unroll
        for (int j = 0; j < 4; ++j) {
          b[j].u[0] = *(const v4u*)(bb + j * 16 * B_STRIDE_Q + ks * 32);
          b[j].u[1] = *(const v4u*)(bb + j * 16 * B_STRIDE_Q + ks * 32 + 16);
        }
#pragma unroll
        for (int rt = 0; rt < 2; ++rt)
#pragma unroll
          for (int j = 0; j < 4; ++j)
            acc[rt][j] = __builtin_amdgcn_wmma_f32_16x16x32_bf16(
                false, a[rt].bf, false, b[j].bf, (short)0, acc[rt][j], false,
                false);
      }
    }

    // Update running argmax; ascending code order + strict '>' keeps
    // first-occurrence semantics like jnp.argmax.
#pragma unroll
    for (int j = 0; j < 4; ++j) {
      const int code = codeBase + j * 16 + ln;
#pragma unroll
      for (int r = 0; r < 8; ++r) {
        if (acc[0][j][r] > bestv0[r]) { bestv0[r] = acc[0][j][r]; besti0[r] = code; }
        if (acc[1][j][r] > bestv1[r]) { bestv1[r] = acc[1][j][r]; besti1[r] = code; }
      }
    }
  }

  // Cross-lane argmax within each 16-lane half (row m = r + hl*8 of a tile).
#pragma unroll
  for (int rt = 0; rt < 2; ++rt) {
    float* bv = rt ? bestv1 : bestv0;
    int*   bi = rt ? besti1 : besti0;
#pragma unroll
    for (int r = 0; r < 8; ++r) {
      float v = bv[r];
      int   i = bi[r];
#pragma unroll
      for (int off = 8; off >= 1; off >>= 1) {
        float ov = __shfl_xor(v, off, 16);
        int   oi = __shfl_xor(i, off, 16);
        if (ov > v || (ov == v && oi < i)) { v = ov; i = oi; }
      }
      if (ln == 0) {
        Val[wave * 32 + rt * 16 + hl * 8 + r] = v;
        Idx[wave * 32 + rt * 16 + hl * 8 + r] = i;
      }
    }
  }
  __syncthreads();

  {
    v2f p;
    p.x = Val[t];
    p.y = __int_as_float(Idx[t]);
    partial[(size_t)(rowBase + t) * CODE_CHUNKS + blockIdx.y] = p;
  }
}

// ------------------- kernel 4: combine chunks + gather ---------------------
__global__ __launch_bounds__(128) void vq_gather(
    const v2f* __restrict__ partial, const float* __restrict__ cb,
    float* __restrict__ out) {
  __shared__ int sIdx;
  const int row = blockIdx.x;
  if (threadIdx.x == 0) {
    float bv = -3.402823466e38f;
    int   bi = 0;
    for (int c = 0; c < CODE_CHUNKS; ++c) {
      v2f p = partial[(size_t)row * CODE_CHUNKS + c];
      int i = __float_as_int(p.y);
      if (p.x > bv || (p.x == bv && i < bi)) { bv = p.x; bi = i; }
    }
    sIdx = bi;
  }
  __syncthreads();
  const v4f* src = (const v4f*)(cb + (size_t)sIdx * DIM);
  v4f*       dst = (v4f*)(out + (size_t)row * DIM);
  dst[threadIdx.x] = src[threadIdx.x];   // 128 threads x 16B = 2KB row
}

// ---------------------------------------------------------------------------
extern "C" void kernel_launch(void* const* d_in, const int* in_sizes, int n_in,
                              void* d_out, int out_size, void* d_ws, size_t ws_size,
                              hipStream_t stream) {
  const float* x  = (const float*)d_in[0];   // [8192, 512]
  const float* cb = (const float*)d_in[1];   // [16384, 512]
  float* out = (float*)d_out;                // [8192, 512]

  unsigned char* ws = (unsigned char*)d_ws;
  unsigned short* xq  = (unsigned short*)ws;                                   // 8 MB
  unsigned short* cbq = (unsigned short*)(ws + (size_t)N_ROWS * DIM * 2);      // 16 MB
  v2f* partial = (v2f*)(ws + (size_t)N_ROWS * DIM * 2 + (size_t)K_CODES * DIM * 2);

  vq_normalize_bf16<<<N_ROWS, 256, 0, stream>>>(x, xq);
  vq_cvt_bf16<<<(K_CODES * DIM) / (256 * 4), 256, 0, stream>>>(cb, cbq);

  dim3 g2(N_ROWS / ROWS_PER_BLOCK, CODE_CHUNKS);
  const size_t smemBytes =
      (size_t)ROWS_PER_BLOCK * LDS_STRIDE * 2 +
      (size_t)2 * 64 * B_STRIDE_Q * 2 +
      (size_t)ROWS_PER_BLOCK * 8;
  vq_sim_argmax<<<g2, 256, smemBytes, stream>>>(xq, cbq, partial);

  vq_gather<<<N_ROWS, 128, 0, stream>>>(partial, cb, out);
}
